// EdgeAttrs_47699906789820
// MI455X (gfx1250) — matrix operand order repaired and checked by reading, not compiled, not run on hardware
//
#include <hip/hip_runtime.h>
#include <hip/hip_bf16.h>

typedef __attribute__((ext_vector_type(16))) __bf16 v16bf;
typedef __attribute__((ext_vector_type(8)))  __bf16 v8bf;
typedef __attribute__((ext_vector_type(4)))  __bf16 v4bf;
typedef __attribute__((ext_vector_type(8)))  float  v8f;
typedef __attribute__((ext_vector_type(4)))  float  v4f;

#define DD 256
#define OO 128
#define PP 32
#define MT 32      // edges per workgroup
#define XS 264     // LDS stride for 256-wide bf16 rows (+8 pad)
#define YS 136     // LDS stride for 128-wide bf16 rows (+8 pad)
#define AS 40      // LDS stride for 32-wide attr rows (+8 pad)

__device__ __forceinline__ v8f wmma_bf16(v16bf a, v16bf b, v8f c) {
    return __builtin_amdgcn_wmma_f32_16x16x32_bf16(false, a, false, b, (short)0, c,
                                                   false, false);
}

// A-fragment (16x32 bf16, MxK) from a row-major LDS buffer.
// Lane L: row = mtile*16 + (L&15); lanes 0-15 hold K {0..7,16..23}, lanes 16-31 hold K {8..15,24..31}.
__device__ __forceinline__ v16bf load_a_lds(const __bf16* base, int stride, int mtile,
                                            int k0, int lane) {
    int m    = mtile * 16 + (lane & 15);
    int koff = (lane >> 4) << 3;
    const __bf16* p = base + m * stride + k0 + koff;
    v8bf lo = *(const v8bf*)(p);
    v8bf hi = *(const v8bf*)(p + 16);
    return __builtin_shufflevector(lo, hi, 0,1,2,3,4,5,6,7,8,9,10,11,12,13,14,15);
}

// B-fragment from the pre-staged layout: 32 lanes x 16 contiguous bf16 per fragment.
__device__ __forceinline__ v16bf load_b_frag(const __bf16* staged, int frag, int lane) {
    return *(const v16bf*)(staged + (((size_t)frag * 32 + lane) << 4));
}

// Restage a row-major f32 weight [K,N] into per-lane WMMA B fragments (bf16).
// Fragment (kblk,nblk): lane L holds column nblk*16+(L&15), K = kblk*32 + (L>>4)*8 + (j&7) + (j>>3)*16, j=0..15.
__global__ void stage_weight(const float* __restrict__ W, int K, int N,
                             __bf16* __restrict__ dst) {
    int e = blockIdx.x * blockDim.x + threadIdx.x;
    if (e >= K * N) return;
    int nblks = N >> 4;
    int f = e >> 9;
    int L = (e >> 4) & 31;
    int j = e & 15;
    int kblk = f / nblks;
    int nblk = f - kblk * nblks;
    int n = nblk * 16 + (L & 15);
    int k = kblk * 32 + ((L >> 4) << 3) + (j & 7) + ((j >> 3) << 4);
    dst[e] = (__bf16)W[k * N + n];
}

__global__ __launch_bounds__(256)
void edge_mlp_kernel(const float* __restrict__ x,
                     const long long* __restrict__ ei,
                     const float* __restrict__ eattr,
                     const __bf16* __restrict__ st,
                     const float* ba0, const float* ba1, const float* ba2, const float* ba3,
                     const float* bb0, const float* bb1, const float* bb2, const float* bb3,
                     const float* __restrict__ Wf,
                     const float* __restrict__ bfin,
                     float* __restrict__ out,
                     int E) {
    __shared__ __align__(32) __bf16 sX[4][MT * XS];   // src, dst, diff, prod (bf16)
    __shared__ __align__(32) __bf16 sH[MT * XS];      // hidden (reused per MLP)
    __shared__ __align__(32) __bf16 sY[4][MT * YS];   // MLP outputs
    __shared__ __align__(32) __bf16 sA[MT * AS];      // edge attr (bf16)
    __shared__ float sDot[MT], sNs[MT], sNd[MT], sCos[MT];
    __shared__ int   sSrc[MT], sDst[MT];

    const int tid  = threadIdx.x;
    const int lane = tid & 31;
    const int w    = tid >> 5;
    const int e0   = blockIdx.x * MT;

    if (tid < MT) {
        sSrc[tid] = (int)ei[e0 + tid];
        sDst[tid] = (int)ei[E + e0 + tid];
        sDot[tid] = 0.f; sNs[tid] = 0.f; sNd[tid] = 0.f;
    }
    __syncthreads();

    // Gather src/dst rows, build X1..X4 in bf16, accumulate cosine partials via LDS atomics.
    for (int it = 0; it < 8; ++it) {
        int q  = tid + it * 256;       // 0..2047 : 32 rows x 64 float4 chunks
        int m  = q >> 6;
        int c4 = q & 63;
        const v4f s4 = *(const v4f*)(x + (size_t)sSrc[m] * DD + c4 * 4);
        const v4f d4 = *(const v4f*)(x + (size_t)sDst[m] * DD + c4 * 4);
        v4bf x1, x2, x3, x4;
        float pd = 0.f, pns = 0.f, pnd = 0.f;
        for (int j = 0; j < 4; ++j) {
            float sv = s4[j], dv = d4[j];
            x1[j] = (__bf16)sv;
            x2[j] = (__bf16)dv;
            x3[j] = (__bf16)(sv - dv);
            x4[j] = (__bf16)(sv * dv);
            pd += sv * dv; pns += sv * sv; pnd += dv * dv;
        }
        int off = m * XS + c4 * 4;
        *(v4bf*)&sX[0][off] = x1;
        *(v4bf*)&sX[1][off] = x2;
        *(v4bf*)&sX[2][off] = x3;
        *(v4bf*)&sX[3][off] = x4;
        atomicAdd(&sDot[m], pd);
        atomicAdd(&sNs[m], pns);
        atomicAdd(&sNd[m], pnd);
    }
    // edge_attr -> bf16 LDS
    for (int it = 0; it < 4; ++it) {
        int q = tid + it * 256;        // 0..1023
        int m = q >> 5, c = q & 31;
        sA[m * AS + c] = (__bf16)eattr[(size_t)(e0 + m) * PP + c];
    }
    __syncthreads();
    if (tid < MT) {
        float ns = fmaxf(sqrtf(sNs[tid]), 1e-8f);
        float nd = fmaxf(sqrtf(sNd[tid]), 1e-8f);
        sCos[tid] = sDot[tid] / (ns * nd);
    }

    const float* bas[4] = {ba0, ba1, ba2, ba3};
    const float* bbs[4] = {bb0, bb1, bb2, bb3};
    const int mt = w >> 2;
    const int lm = ((lane >> 4) << 3);

    for (int i = 0; i < 4; ++i) {
        const __bf16* stWa = st + (size_t)i * 65536;
        const __bf16* stWb = st + 262144 + (size_t)i * 32768;
        __syncthreads();  // sH free from previous MLP; sX ready (and sCos on i==0)

        // layer 1: H = relu(X_i @ Wa + ba)  : 32 tiles, 4 per wave
        {
            const int ntb = (w & 3) * 4;
            v8f acc[4] = {};
            for (int kb = 0; kb < 8; ++kb) {
                v16bf a = load_a_lds(&sX[i][0], XS, mt, kb * 32, lane);
                for (int j = 0; j < 4; ++j)
                    acc[j] = wmma_bf16(a, load_b_frag(stWa, kb * 16 + ntb + j, lane), acc[j]);
            }
            const float* ba = bas[i];
            for (int j = 0; j < 4; ++j) {
                int n = (ntb + j) * 16 + (lane & 15);
                float bv = ba[n];
                for (int r = 0; r < 8; ++r) {
                    int m = mt * 16 + r + lm;
                    sH[m * XS + n] = (__bf16)fmaxf(acc[j][r] + bv, 0.f);
                }
            }
        }
        __syncthreads();

        // layer 2: Y_i = relu(H @ Wb + bb)  : 16 tiles, 2 per wave
        {
            const int ntb = (w & 3) * 2;
            v8f acc[2] = {};
            for (int kb = 0; kb < 8; ++kb) {
                v16bf a = load_a_lds(sH, XS, mt, kb * 32, lane);
                for (int j = 0; j < 2; ++j)
                    acc[j] = wmma_bf16(a, load_b_frag(stWb, kb * 8 + ntb + j, lane), acc[j]);
            }
            const float* bb = bbs[i];
            for (int j = 0; j < 2; ++j) {
                int n = (ntb + j) * 16 + (lane & 15);
                float bv = bb[n];
                for (int r = 0; r < 8; ++r) {
                    int m = mt * 16 + r + lm;
                    sY[i][m * YS + n] = (__bf16)fmaxf(acc[j][r] + bv, 0.f);
                }
            }
        }
    }
    __syncthreads();

    // Head: out = tanh([Y1..Y4, s, attr] @ Wf + bf)
    {
        const int ntb = (w & 3) * 2;
        v8f acc[2] = {};
        for (int p = 0; p < 4; ++p) {
            const __bf16* stWfp = st + 393216 + (size_t)p * 16384;
            for (int kb = 0; kb < 4; ++kb) {
                v16bf a = load_a_lds(&sY[p][0], YS, mt, kb * 32, lane);
                for (int j = 0; j < 2; ++j)
                    acc[j] = wmma_bf16(a, load_b_frag(stWfp, kb * 8 + ntb + j, lane), acc[j]);
            }
        }
        {
            const __bf16* stWfa = st + 458752;  // attr rows 513..544 of Wf
            v16bf a = load_a_lds(sA, AS, mt, 0, lane);
            for (int j = 0; j < 2; ++j)
                acc[j] = wmma_bf16(a, load_b_frag(stWfa, ntb + j, lane), acc[j]);
        }
        for (int j = 0; j < 2; ++j) {
            int n = (ntb + j) * 16 + (lane & 15);
            float wsn = Wf[512 * OO + n];   // cosine-sim row of Wf
            float bv  = bfin[n];
            for (int r = 0; r < 8; ++r) {
                int m = mt * 16 + r + lm;
                float v = acc[j][r] + sCos[m] * wsn + bv;
                out[(size_t)(e0 + m) * OO + n] = tanhf(v);
            }
        }
    }
}

extern "C" void kernel_launch(void* const* d_in, const int* in_sizes, int n_in,
                              void* d_out, int out_size, void* d_ws, size_t ws_size,
                              hipStream_t stream) {
    (void)n_in; (void)out_size; (void)ws_size;
    const float*     x   = (const float*)d_in[0];
    const long long* ei  = (const long long*)d_in[1];
    const float*     ea  = (const float*)d_in[2];
    const float* W1a = (const float*)d_in[3];
    const float* b1a = (const float*)d_in[4];
    const float* W1b = (const float*)d_in[5];
    const float* b1b = (const float*)d_in[6];
    const float* W2a = (const float*)d_in[7];
    const float* b2a = (const float*)d_in[8];
    const float* W2b = (const float*)d_in[9];
    const float* b2b = (const float*)d_in[10];
    const float* W3a = (const float*)d_in[11];
    const float* b3a = (const float*)d_in[12];
    const float* W3b = (const float*)d_in[13];
    const float* b3b = (const float*)d_in[14];
    const float* W4a = (const float*)d_in[15];
    const float* b4a = (const float*)d_in[16];
    const float* W4b = (const float*)d_in[17];
    const float* b4b = (const float*)d_in[18];
    const float* Wf  = (const float*)d_in[19];
    const float* bf  = (const float*)d_in[20];

    const int E = in_sizes[1] / 2;     // edge_index is [2, E]
    __bf16* st = (__bf16*)d_ws;        // ~904 KB of staged bf16 weight fragments

    auto stage = [&](const float* W, int K, int N, size_t off) {
        int total = K * N;
        stage_weight<<<(total + 255) / 256, 256, 0, stream>>>(W, K, N, st + off);
    };
    stage(W1a, 256, 256, 0);
    stage(W2a, 256, 256, 65536);
    stage(W3a, 256, 256, 131072);
    stage(W4a, 256, 256, 196608);
    stage(W1b, 256, 128, 262144);
    stage(W2b, 256, 128, 262144 + 32768);
    stage(W3b, 256, 128, 262144 + 65536);
    stage(W4b, 256, 128, 262144 + 98304);
    stage(Wf + 0 * 128 * 128, 128, 128, 393216);
    stage(Wf + 1 * 128 * 128, 128, 128, 393216 + 16384);
    stage(Wf + 2 * 128 * 128, 128, 128, 393216 + 32768);
    stage(Wf + 3 * 128 * 128, 128, 128, 393216 + 49152);
    stage(Wf + 513 * 128, 32, 128, 458752);

    edge_mlp_kernel<<<E / MT, 256, 0, stream>>>(
        x, ei, ea, st,
        b1a, b2a, b3a, b4a, b1b, b2b, b3b, b4b,
        Wf, bf, (float*)d_out, E);
}
